// QKVAttentionLegacy_81458349736746
// MI455X (gfx1250) — matrix operand order repaired and checked by reading, not compiled, not run on hardware
//
#include <hip/hip_runtime.h>

typedef __attribute__((ext_vector_type(16))) _Float16 v16h;
typedef __attribute__((ext_vector_type(4)))  _Float16 v4h;
typedef __attribute__((ext_vector_type(8)))  float    v8f;

#define T_LEN 2048
#define CH    64
#define SC    64          // s-chunk per iteration
#define TB    128         // t-rows per block (8 waves x 16)

// All scaling folded into Q:  logit2 = (q . k / 8) * log2(e)
#define QSCALE 0.18033688011112042f   // log2(e)/8
// Fixed base-2 shift (softmax is shift-invariant; exact):
#define C_SHIFT (-5.7707801635558535f) // -4*log2(e); P = 2^(logit2+shift) in [2^-13, 32]

// qkv: (4, 3072, 2048) fp32 = 64 heads x [q(64xT), k(64xT), v(64xT)]
// out: (4, 1024, 2048) fp32 = per head 64 x T
__global__ __launch_bounds__(256)
void qkv_attn_fa_kernel(const float* __restrict__ qkv, float* __restrict__ out) {
    const int head = blockIdx.x >> 4;        // 0..63
    const int t0   = (blockIdx.x & 15) * TB; // t-tile base
    const int tid  = threadIdx.x;            // 0..255
    const int wave = tid >> 5;               // 0..7
    const int lane = tid & 31;
    const int half = lane >> 4;
    const int l16  = lane & 15;

    const float* qbase = qkv + (size_t)head * 192 * T_LEN;
    const float* kbase = qbase + (size_t)64  * T_LEN;
    const float* vbase = qbase + (size_t)128 * T_LEN;

    // LDS: loop phase   K frags 8KB | V frags 8KB | P frags 16KB (2KB/wave)
    //      final phase  reused as fp32 O tile 64 x 132
    __shared__ __align__(16) unsigned char smem[64 * 132 * 4];
    _Float16* ldsK = (_Float16*)smem;            // 8 frags * 32 lanes * 16 halves
    _Float16* ldsV = (_Float16*)(smem + 8192);   // 8 frags * 32 * 16
    _Float16* ldsP = (_Float16*)(smem + 16384);  // 8 waves * 2 frags * 32 * 16

    // ---- Q A-fragments (16x32 f16), loaded once, fully pre-scaled ----
    // A layout: lane m=l16, kb=half*8; elem e<8 -> k=kb+e ; elem 8+e -> k=16+kb+e
    v16h qa[2];
    {
        const int trow = t0 + wave * 16 + l16;
        const int kb = half * 8;
#pragma unroll
        for (int kc = 0; kc < 2; ++kc) {
#pragma unroll
            for (int e = 0; e < 8; ++e) {
                qa[kc][e]     = (_Float16)(qbase[(size_t)(kc * 32 + kb + e)      * T_LEN + trow] * QSCALE);
                qa[kc][e + 8] = (_Float16)(qbase[(size_t)(kc * 32 + 16 + kb + e) * T_LEN + trow] * QSCALE);
            }
        }
    }

    v8f o[4] = {};    // O accumulators: 4 ch-tiles of 16
    v8f lacc = {};    // row sums of P via ones-WMMA, accumulated over all chunks
    v8f cinit;        // score-accumulator init = base-2 shift (free: WMMA SRC2 != VDST)
    v16h vones;
#pragma unroll
    for (int r = 0; r < 8; ++r) cinit[r] = C_SHIFT;
#pragma unroll
    for (int e = 0; e < 16; ++e) vones[e] = (_Float16)1.0f;

    for (int s0 = 0; s0 < T_LEN; s0 += SC) {
        __syncthreads();   // previous chunk's fragment reads complete

        // ---- K chunk -> LDS in B-fragment layout (frag = kc*4+j); pure cvt ----
        // value K[c][s] -> frag(kc=c>>5, j=s>>4), lane L=((c>>3)&1)*16+(s&15), elem e=(c&7)+((c>>4)&1)*8
#pragma unroll
        for (int it = 0; it < 4; ++it) {
            int i = tid + it * 256;
            int s = i & 63, c0 = (i >> 6) << 2;            // pack 4 consecutive c
            int kc = c0 >> 5, hb = (c0 >> 3) & 1, e0 = (c0 & 7) + ((c0 >> 4) & 1) * 8;
            int L = hb * 16 + (s & 15), j = s >> 4;
            v4h pk;
#pragma unroll
            for (int d = 0; d < 4; ++d)
                pk[d] = (_Float16)kbase[(size_t)(c0 + d) * T_LEN + s0 + s];
            *(v4h*)(ldsK + ((kc * 4 + j) * 32 + L) * 16 + e0) = pk;
        }
        // ---- V chunk -> LDS in B-fragment layout (frag = kc2*4+nt) ----
        // value V[c][s] -> frag(kc2=s>>5, nt=c>>4), lane L=((s>>3)&1)*16+(c&15), elem e=(s&7)+((s>>4)&1)*8
#pragma unroll
        for (int it = 0; it < 4; ++it) {
            int i = tid + it * 256;
            int c = i >> 4, s = (i & 15) << 2;             // pack 4 consecutive s (float4)
            int kc2 = s >> 5, hb = (s >> 3) & 1, e0 = (s & 7) + ((s >> 4) & 1) * 8;
            int L = hb * 16 + (c & 15), nt = c >> 4;
            const float4 f = *(const float4*)(vbase + (size_t)c * T_LEN + s0 + s);
            v4h pk;
            pk[0] = (_Float16)f.x; pk[1] = (_Float16)f.y;
            pk[2] = (_Float16)f.z; pk[3] = (_Float16)f.w;
            *(v4h*)(ldsV + ((kc2 * 4 + nt) * 32 + L) * 16 + e0) = pk;
        }
        __syncthreads();

        // ---- scores: 16(t) x 64(s); C pre-loaded with shift constant ----
        v8f sc[4];
#pragma unroll
        for (int j = 0; j < 4; ++j) {
            v8f acc = cinit;
#pragma unroll
            for (int kc = 0; kc < 2; ++kc) {
                v16h bf = *(const v16h*)(ldsK + ((kc * 4 + j) * 32 + lane) * 16);
                acc = __builtin_amdgcn_wmma_f32_16x16x32_f16(
                        false, qa[kc], false, bf, (short)0, acc, false, false);
            }
            sc[j] = acc;
        }

        // ---- P = 2^(shifted logit): one v_exp_f32 each; store in A-frag layout ----
        _Float16* Pw = ldsP + wave * 1024;
#pragma unroll
        for (int j = 0; j < 4; ++j) {
            const int n   = j * 16 + l16;       // s within chunk
            const int kl  = n & 31;
            const int kc2 = n >> 5;
            const int lrHi = ((kl >> 3) & 1) * 16;
            const int e    = (kl & 7) + ((kl >> 4) & 1) * 8;
#pragma unroll
            for (int r = 0; r < 8; ++r) {
                float p = __builtin_amdgcn_exp2f(sc[j][r]);
                int m = r + half * 8;
                Pw[(kc2 * 32 + (m + lrHi)) * 16 + e] = (_Float16)p;
            }
        }
        // same-wave DS is in-order; read A-fragments back contiguously
        v16h pa[2];
#pragma unroll
        for (int kc2 = 0; kc2 < 2; ++kc2)
            pa[kc2] = *(const v16h*)(Pw + (kc2 * 32 + lane) * 16);

        // ---- O += P(16x64) x V^T ; row sums via ones-column WMMA ----
#pragma unroll
        for (int kc2 = 0; kc2 < 2; ++kc2) {
            lacc = __builtin_amdgcn_wmma_f32_16x16x32_f16(
                     false, pa[kc2], false, vones, (short)0, lacc, false, false);
#pragma unroll
            for (int nt = 0; nt < 4; ++nt) {
                v16h bv = *(const v16h*)(ldsV + ((kc2 * 4 + nt) * 32 + lane) * 16);
                o[nt] = __builtin_amdgcn_wmma_f32_16x16x32_f16(
                          false, pa[kc2], false, bv, (short)0, o[nt], false, false);
            }
        }
    }

    // ---- normalize, transpose through LDS, coalesced float4 stores ----
    __syncthreads();
    float* ldsO = (float*)smem;   // 64 c-rows x 132 (128 t + pad)
#pragma unroll
    for (int r = 0; r < 8; ++r) {
        float inv = 1.0f / lacc[r];
        int tloc = wave * 16 + r + half * 8;
#pragma unroll
        for (int nt = 0; nt < 4; ++nt)
            ldsO[(nt * 16 + l16) * 132 + tloc] = o[nt][r] * inv;
    }
    __syncthreads();
    float* obase = out + (size_t)head * CH * T_LEN;
#pragma unroll
    for (int it = 0; it < 8; ++it) {
        int i = tid + it * 256;
        int t4 = (i & 31) << 2, c = i >> 5;
        float4 val = *(const float4*)(ldsO + c * 132 + t4);
        *(float4*)(obase + (size_t)c * T_LEN + t0 + t4) = val;
    }
}

extern "C" void kernel_launch(void* const* d_in, const int* in_sizes, int n_in,
                              void* d_out, int out_size, void* d_ws, size_t ws_size,
                              hipStream_t stream) {
    const float* qkv = (const float*)d_in[0];
    float* out = (float*)d_out;
    // 64 heads * 16 t-tiles of 128 rows; 8 waves (256 threads) per block
    dim3 grid(64 * 16);
    dim3 block(256);
    qkv_attn_fa_kernel<<<grid, block, 0, stream>>>(qkv, out);
}
